// GruKRegion_59837484368091
// MI455X (gfx1250) — matrix operand order repaired and checked by reading, not compiled, not run on hardware
//
#include <hip/hip_runtime.h>
#include <hip/hip_bf16.h>

// ---------------------------------------------------------------------------
// Bidirectional stacked GRU (B=32, D=400, K=200, H=8, 4 layers) for gfx1250.
//
// One wave32 per 16-sequence block per direction. Each timestep:
//   2x v_wmma_f32_16x16x32_f16   (r/z combined tile, inn|hn split tile)
//   gate math with v_exp_f32 / v_rcp_f32 fast paths (no libm, no divides)
//   cross-column exchange via DPP16 ROW_ROR:8 (xor-8) - no ds_bpermute
//   8x16 f16 LDS transpose back to sequence domain (single wave => DS ops
//   are in-order, no barriers needed)
// Both 82MB ping-pong activation buffers live in the 192MB L2.
// ---------------------------------------------------------------------------

typedef __attribute__((ext_vector_type(16))) _Float16 v16h;
typedef __attribute__((ext_vector_type(8)))  float    v8f;
typedef unsigned int u32;

static constexpr int Bn = 32, Dn = 400, Kn = 200, Hn = 8, OUTn = 8;
static constexpr int NSEQ = Bn * Kn;        // 6400 sequences
static constexpr int NBLK = NSEQ / 16;      // 400 blocks of 16 sequences
static constexpr size_t BUFELEMS = (size_t)NBLK * Dn * 16 * 16; // f16 elems per buffer
static constexpr size_t BUFBYTES = BUFELEMS * 2;                // 81,920,000 B

union V16H { v16h v; u32 u[8]; };

__device__ __forceinline__ u32 packh2(float a, float b) {
  union { _Float16 h[2]; u32 u; } q;
  q.h[0] = (_Float16)a; q.h[1] = (_Float16)b;
  return q.u;
}
__device__ __forceinline__ int imax(int a, int b) { return a > b ? a : b; }
__device__ __forceinline__ int imin(int a, int b) { return a < b ? a : b; }

// xor-8 lane exchange within wave32 == DPP16 ROW_ROR:8 (self-inverse).
__device__ __forceinline__ float dpp_xor8(float v) {
  int i = __builtin_bit_cast(int, v);
  int r = __builtin_amdgcn_update_dpp(0, i, 0x128 /*ROW_ROR:8*/, 0xf, 0xf, true);
  return __builtin_bit_cast(float, r);
}

// Fast transcendentals: v_exp_f32 + v_rcp_f32, no IEEE divide, no branches.
__device__ __forceinline__ float fast_rcp(float x) { return __builtin_amdgcn_rcpf(x); }
__device__ __forceinline__ float fast_sigmoid(float x) {
  return fast_rcp(1.0f + __expf(-x));
}
__device__ __forceinline__ float fast_tanh(float x) {
  float x2 = fminf(fmaxf(2.0f * x, -30.0f), 30.0f);   // med3, avoids exp overflow
  float e  = __expf(x2);
  return (e - 1.0f) * fast_rcp(e + 1.0f);
}

// ---------------------------------------------------------------------------
// Kernel 1: per-sequence length = index of first exact 0.0 (else D).
// seq[n, t] = x[b, t, k], n = b*K + k.
// ---------------------------------------------------------------------------
__global__ void k_len(const float* __restrict__ x, int* __restrict__ len) {
  int n = blockIdx.x * blockDim.x + threadIdx.x;
  if (n >= NSEQ) return;
  int b = n / Kn, k = n % Kn;
  const float* p = x + (size_t)b * Dn * Kn + k;
  int L = Dn;
  for (int t = 0; t < Dn; ++t) {
    if (p[(size_t)t * Kn] == 0.0f) { L = t; break; }
  }
  len[n] = L;
}

// ---------------------------------------------------------------------------
// Weight accessors (layer 0 has input dim 1, padded to 16 with zeros).
// ---------------------------------------------------------------------------
__device__ __forceinline__ float getWih(const float* w0, const float* wl,
                                        int l, int dir, int g, int i) {
  if (l == 0) return (i == 0) ? w0[dir * 3 * Hn + g] : 0.0f;       // (2,24,1)
  return wl[((((l - 1) * 2 + dir) * 3 * Hn) + g) * (2 * Hn) + i];  // (3,2,24,16)
}
__device__ __forceinline__ float getWhh(const float* w0, const float* wl,
                                        int l, int dir, int g, int u) {
  if (l == 0) return w0[(dir * 3 * Hn + g) * Hn + u];              // (2,24,8)
  return wl[((((l - 1) * 2 + dir) * 3 * Hn) + g) * Hn + u];        // (3,2,24,8)
}
__device__ __forceinline__ float getBias(const float* b0, const float* bl,
                                         int l, int dir, int g) {
  if (l == 0) return b0[dir * 3 * Hn + g];
  return bl[(((l - 1) * 2 + dir) * 3 * Hn) + g];
}

// B-tile element B[k][j] for tile 0 (r,z combined) / tile 1 (inn | hn split).
__device__ __forceinline__ float btileElem(
    const float* wih0, const float* wih, const float* whh0, const float* whh,
    int l, int dir, int tile, int k, int j) {
  if (tile == 0) {                       // cols j=0..7 r-gate, 8..15 z-gate
    if (k < 16) return getWih(wih0, wih, l, dir, j, k);
    if (k < 24) return getWhh(whh0, whh, l, dir, j, k - 16);
    return 0.0f;
  } else {
    if (j < 8)                           // inn_j from x rows only
      return (k < 16) ? getWih(wih0, wih, l, dir, 16 + j, k) : 0.0f;
    // hn_{j-8} from h rows only (Whh n-rows are 16..23)
    return (k >= 16 && k < 24) ? getWhh(whh0, whh, l, dir, 8 + j, k - 16) : 0.0f;
  }
}

// ---------------------------------------------------------------------------
// Kernel 2: pack B-tiles into the 16-bit B-operand (32x16) VGPR layout:
// VGPR v, lane l: N = l%16, K = 2v + (v<4 ? 0 : 8) + (l<16 ? 0 : 8), f16 pair [K, K+1].
// Fold biases per gate column: bias1[j]=bih[j]+bhh[j] (r,z);
// bias2[j<8]=bih[16+j] (inn), bias2[j>=8]=bhh[8+j] (hn).
// ---------------------------------------------------------------------------
__global__ void k_prep_w(const float* __restrict__ wih0, const float* __restrict__ whh0,
                         const float* __restrict__ bih0, const float* __restrict__ bhh0,
                         const float* __restrict__ wih,  const float* __restrict__ whh,
                         const float* __restrict__ bih,  const float* __restrict__ bhh,
                         u32* __restrict__ btile, float* __restrict__ bias) {
  int tid = threadIdx.x;                      // 512 threads
  int l    = tid >> 7;
  int dir  = (tid >> 6) & 1;
  int tile = (tid >> 5) & 1;
  int lane = tid & 31;
  int N = lane & 15;
  int kofs = (lane < 16) ? 0 : 8;
#pragma unroll
  for (int v = 0; v < 8; ++v) {
    int K0 = 2 * v + (v < 4 ? 0 : 8) + kofs;
    float a = btileElem(wih0, wih, whh0, whh, l, dir, tile, K0,     N);
    float b = btileElem(wih0, wih, whh0, whh, l, dir, tile, K0 + 1, N);
    btile[(((l * 2 + dir) * 2 + tile) * 32 + lane) * 8 + v] = packh2(a, b);
  }
  if (tid < 256) {                            // biases: tid = l*64+dir*32+which*16+j
    int bl    = tid >> 6;
    int bdir  = (tid >> 5) & 1;
    int which = (tid >> 4) & 1;
    int j     = tid & 15;
    float val;
    if (which == 0)
      val = getBias(bih0, bih, bl, bdir, j) + getBias(bhh0, bhh, bl, bdir, j);
    else
      val = (j < 8) ? getBias(bih0, bih, bl, bdir, 16 + j)
                    : getBias(bhh0, bhh, bl, bdir, 8 + j);
    bias[tid] = val;
  }
}

// ---------------------------------------------------------------------------
// Kernel 3: layer-0 input buffer, f16, A-operand-ready layout:
//   buf[blk][t][m][dim], dim0 = x value, dims 1..15 = 0.
// ---------------------------------------------------------------------------
__global__ void k_prep_input(const float* __restrict__ x, _Float16* __restrict__ buf) {
  int idx = blockIdx.x * blockDim.x + threadIdx.x;
  if (idx >= NSEQ * Dn) return;
  int n = idx / Dn, t = idx % Dn;
  int b = n / Kn, k = n % Kn;
  float val = x[((size_t)b * Dn + t) * Kn + k];
  _Float16* dst = buf + (((size_t)(n >> 4) * Dn + t) * 16 + (n & 15)) * 16;
  uint4 first = { packh2(val, 0.0f), 0u, 0u, 0u };
  uint4 zero4 = { 0u, 0u, 0u, 0u };
  *(uint4*)dst       = first;
  *(uint4*)(dst + 8) = zero4;
}

// ---------------------------------------------------------------------------
// Kernel 4: the recurrent GRU scan. One wave32 per (16-seq block, direction).
// ---------------------------------------------------------------------------
__global__ __launch_bounds__(32)
void k_scan(const _Float16* __restrict__ in, _Float16* __restrict__ out,
            const u32* __restrict__ btile, const float* __restrict__ bias,
            const int* __restrict__ len, int layer) {
  const int blk  = blockIdx.x;
  const int dir  = blockIdx.y;          // 0 = fwd, 1 = bwd
  const int lane = threadIdx.x;
  const int m  = lane & 15;             // sequence (loads/stores) / gate column
  const int hi = lane >> 4;

  __shared__ _Float16 ldsT[16 * 8];     // [seq 0..15][unit 0..7] transpose tile
  __shared__ int ldsLen[16];

  const int lenM = len[blk * 16 + m];
  if (lane < 16) ldsLen[lane] = lenM;
  // Single wave: DS ops are in-order, no barrier needed (compiler adds dscnt wait).

  // Rows handled by this lane in the D tile: M = r + 8*hi.
  int lenRow[8];
#pragma unroll
  for (int r = 0; r < 8; ++r) lenRow[r] = ldsLen[r + 8 * hi];

  // Tmax over the block (wave-wide max reduce).
  int tm = (lane < 16) ? lenM : 0;
#pragma unroll
  for (int s = 16; s >= 1; s >>= 1) tm = imax(tm, __shfl_xor(tm, s, 32));
  const int Tmax = tm;

  // Per-(layer,dir) constant B tiles and column biases.
  const u32* bt = btile + (size_t)((layer * 2 + dir) * 2) * 256;
  V16H B1, B2;
#pragma unroll
  for (int v = 0; v < 8; ++v) {
    B1.u[v] = bt[lane * 8 + v];
    B2.u[v] = bt[256 + lane * 8 + v];
  }
  const float* bs = bias + (layer * 2 + dir) * 32;
  const float bias1 = bs[m];
  const float bias2 = bs[16 + m];

  float hg[8];                          // gate-domain hidden (valid on cols<8)
#pragma unroll
  for (int r = 0; r < 8; ++r) hg[r] = 0.0f;
  uint4 hseq = { 0u, 0u, 0u, 0u };      // sequence-domain hidden (A VGPRs 4..7)

  const size_t baseIn = (size_t)blk * Dn * 256;  // f16 elems per block row
  const _Float16* inb = in + baseIn + (size_t)m * 16 + 8 * hi;

  // Software pipeline: preload t=0, inside loop issue t+1 before gate math.
  const int tr0 = (dir == 0) ? 0 : imax(lenM - 1, 0);
  uint4 xin = (Tmax > 0) ? *(const uint4*)(inb + (size_t)tr0 * 256)
                         : uint4{0u, 0u, 0u, 0u};
  const v8f cz = {};

  for (int t = 0; t < Tmax; ++t) {
    // Issue next timestep's load early (clamped address, always in-bounds).
    const int tn  = imin(t + 1, Tmax - 1);
    const int trn = (dir == 0) ? tn : imax(lenM - 1 - tn, 0);
    uint4 xnext = *(const uint4*)(inb + (size_t)trn * 256);
    {  // warm L0 two steps ahead
      const int tp  = imin(t + 2, Tmax - 1);
      const int trp = (dir == 0) ? tp : imax(lenM - 1 - tp, 0);
      __builtin_prefetch((const void*)(inb + (size_t)trp * 256), 0, 3);
    }

    // A (16x32 f16): lanes<16: K0..7 = x dims0..7, K16..23 = h units0..7
    //                lanes>=16: K8..15 = x dims8..15, K24..31 = 0
    V16H A;
    A.u[0] = xin.x; A.u[1] = xin.y; A.u[2] = xin.z; A.u[3] = xin.w;
    if (hi == 0) { A.u[4] = hseq.x; A.u[5] = hseq.y; A.u[6] = hseq.z; A.u[7] = hseq.w; }
    else         { A.u[4] = 0u; A.u[5] = 0u; A.u[6] = 0u; A.u[7] = 0u; }

    v8f d1 = __builtin_amdgcn_wmma_f32_16x16x32_f16(false, A.v, false, B1.v,
                                                    (short)0, cz, false, false);
    v8f d2 = __builtin_amdgcn_wmma_f32_16x16x32_f16(false, A.v, false, B2.v,
                                                    (short)0, cz, false, false);

    // Gate math. Column j = m: cols 0..7 carry r/inn, cols 8..15 carry z/hn.
    // Each lane sigmoids its own pre-activation (r on cols<8, z on cols>=8),
    // then DPP xor-8 brings z and hn over to the r/n columns.
    float hout[8];
#pragma unroll
    for (int r = 0; r < 8; ++r) {
      float srz = d1[r] + bias1;              // ir+hr (j<8) | iz+hz (j>=8)
      float sg  = fast_sigmoid(srz);          // r (j<8) | z (j>=8)
      float sn  = d2[r] + bias2;              // inn (j<8) | hn (j>=8)
      float hnb = dpp_xor8(sn);               // hn  -> cols<8
      float zg  = dpp_xor8(sg);               // z   -> cols<8
      float ng  = fast_tanh(sn + sg * hnb);   // n on cols<8 (sg there = r)
      float hnew = ng + zg * (hg[r] - ng);    // (1-z)*n + z*h
      bool mt = (t < lenRow[r]);
      hg[r]   = mt ? hnew : hg[r];
      hout[r] = mt ? hg[r] : 0.0f;            // masked output
    }

    // Transpose gate-domain h -> sequence-domain via 256B LDS tile.
    // Same-wave DS ops are in-order: no barriers required.
    if (m < 8) {
#pragma unroll
      for (int r = 0; r < 8; ++r)
        ldsT[(r + 8 * hi) * 8 + m] = (_Float16)hout[r];
    }
    uint4 hs = { 0u, 0u, 0u, 0u };
    if (hi == 0) hs = *(const uint4*)&ldsT[lane * 8];
    hseq = hs;

    // Store layer output (f16, A-ready): fwd -> dims0..7 (zeros masked rows,
    // plus zero-fill dims8..15 where t>=len so bwd-gap positions are clean);
    // bwd -> dims8..15 at original position len-1-t.
    if (hi == 0) {
      if (dir == 0) {
        _Float16* po = out + baseIn + ((size_t)t * 16 + lane) * 16;
        *(uint4*)po = hs;
        if (t >= lenM) {
          uint4 zero4 = { 0u, 0u, 0u, 0u };
          *(uint4*)(po + 8) = zero4;
        }
      } else if (t < lenM) {
        int to = lenM - 1 - t;
        _Float16* po = out + baseIn + ((size_t)to * 16 + lane) * 16 + 8;
        *(uint4*)po = hs;
      }
    }

    xin = xnext;
  }
}

// ---------------------------------------------------------------------------
// Kernel 5: head. last hidden -> lin1 (leaky 0.2) -> lin2, f32 out (N,8).
// ---------------------------------------------------------------------------
__global__ void k_head(const _Float16* __restrict__ fin, const int* __restrict__ len,
                       const float* __restrict__ l1w, const float* __restrict__ l1b,
                       const float* __restrict__ l2w, const float* __restrict__ l2b,
                       float* __restrict__ outp) {
  int n = blockIdx.x * blockDim.x + threadIdx.x;
  if (n >= NSEQ) return;
  int L = len[n];
  int t = (L > 0) ? (L - 1) : 0;
  const _Float16* p = fin + (((size_t)(n >> 4) * Dn + t) * 16 + (n & 15)) * 16;
  float f[16];
#pragma unroll
  for (int i = 0; i < 16; ++i) f[i] = (float)p[i];
  float y[8];
#pragma unroll
  for (int j = 0; j < 8; ++j) {
    float a = l1b[j];
#pragma unroll
    for (int i = 0; i < 16; ++i) a += f[i] * l1w[j * 16 + i];
    y[j] = (a > 0.0f) ? a : 0.2f * a;
  }
#pragma unroll
  for (int o = 0; o < 8; ++o) {
    float a = l2b[o];
#pragma unroll
    for (int j = 0; j < 8; ++j) a += y[j] * l2w[o * 8 + j];
    outp[(size_t)n * 8 + o] = a;
  }
}

// ---------------------------------------------------------------------------
extern "C" void kernel_launch(void* const* d_in, const int* in_sizes, int n_in,
                              void* d_out, int out_size, void* d_ws, size_t ws_size,
                              hipStream_t stream) {
  (void)in_sizes; (void)n_in; (void)out_size; (void)ws_size;
  const float* x     = (const float*)d_in[0];
  const float* w_ih0 = (const float*)d_in[1];
  const float* w_hh0 = (const float*)d_in[2];
  const float* b_ih0 = (const float*)d_in[3];
  const float* b_hh0 = (const float*)d_in[4];
  const float* w_ih  = (const float*)d_in[5];
  const float* w_hh  = (const float*)d_in[6];
  const float* b_ih  = (const float*)d_in[7];
  const float* b_hh  = (const float*)d_in[8];
  const float* l1w   = (const float*)d_in[9];
  const float* l1b   = (const float*)d_in[10];
  const float* l2w   = (const float*)d_in[11];
  const float* l2b   = (const float*)d_in[12];

  char* ws = (char*)d_ws;
  int*      len   = (int*)(ws + 0);                    // 25,600 B
  float*    bias  = (float*)(ws + 32 * 1024);          // 1,024 B
  u32*      btile = (u32*)(ws + 36 * 1024);            // 16,384 B
  _Float16* buf0  = (_Float16*)(ws + 64 * 1024);       // 81.92 MB (L2-resident)
  _Float16* buf1  = (_Float16*)(ws + 64 * 1024 + BUFBYTES);

  k_len<<<(NSEQ + 255) / 256, 256, 0, stream>>>(x, len);
  k_prep_w<<<1, 512, 0, stream>>>(w_ih0, w_hh0, b_ih0, b_hh0,
                                  w_ih, w_hh, b_ih, b_hh, btile, bias);
  k_prep_input<<<(NSEQ * Dn + 255) / 256, 256, 0, stream>>>(x, buf0);

  _Float16* bin = buf0;
  _Float16* bout = buf1;
  for (int l = 0; l < 4; ++l) {
    dim3 grid(NBLK, 2);                  // fwd & bwd directions in parallel
    k_scan<<<grid, 32, 0, stream>>>(bin, bout, btile, bias, len, l);
    _Float16* tmp = bin; bin = bout; bout = tmp;
  }
  // After 4 swaps the final layer's output is back in buf0 (== bin).
  k_head<<<(NSEQ + 255) / 256, 256, 0, stream>>>(bin, len, l1w, l1b, l2w, l2b,
                                                 (float*)d_out);
}
